// SimpleScan_73581379715190
// MI455X (gfx1250) — matrix operand order repaired and checked by reading, not compiled, not run on hardware
//
#include <hip/hip_runtime.h>

// ---------------------------------------------------------------------------
// MI455X (gfx1250, wave32) implementation of conv1d+pool -> 2x LSTM(scan over
// channels) -> dense.  All matrix work runs on v_wmma_f32_16x16x32_bf16.
// Weights are pre-transposed to [n][k] bf16 (K padded to mult of 32) so each
// WMMA B operand is one contiguous 32B load per lane; A operands use the same
// contiguous k order (16 elements at K = kgrp*16 per lane, two adjacent 16B loads).
// ---------------------------------------------------------------------------

typedef __attribute__((ext_vector_type(16))) __bf16 v16bf;
typedef __attribute__((ext_vector_type(8)))  __bf16 v8bf;
typedef __attribute__((ext_vector_type(8)))  float  v8f;
typedef __attribute__((ext_vector_type(4)))  unsigned int u32x4;

#define Bsz   256
#define Lseq  2048
#define CIN   64
#define CO    32
#define LP    1024
#define HID   100
#define NG    400     // 4*HID
#define NCLS  95

__device__ __forceinline__ __bf16 f2bf(float f) {
  union { float f; unsigned int u; } a; a.f = f;
  unsigned int r = (a.u + 0x7FFFu + ((a.u >> 16) & 1u)) >> 16;  // RNE
  union { unsigned short s; __bf16 b; } o; o.s = (unsigned short)r;
  return o.b;
}

__device__ __forceinline__ float sigm(float x) { return 1.f / (1.f + __expf(-x)); }
__device__ __forceinline__ float tanh_(float x) {
  float e = __expf(-2.f * fabsf(x));
  float r = (1.f - e) / (1.f + e);
  return x >= 0.f ? r : -r;
}

// A operand: 16 contiguous elements per lane (K = kgrp*16+[0..15])
__device__ __forceinline__ v16bf load_tileA(const __bf16* p) {
  union { v16bf v; v8bf h[2]; } u;
  u.h[0] = *(const v8bf*)p;
  u.h[1] = *(const v8bf*)(p + 8);
  return u.v;
}
// B operand: 16 contiguous bf16 per lane (K ascending), halves +0 and +8 elements
__device__ __forceinline__ v16bf load_tileB(const __bf16* p) {
  union { v16bf v; v8bf h[2]; } u;
  u.h[0] = *(const v8bf*)p;
  u.h[1] = *(const v8bf*)(p + 8);
  return u.v;
}

__device__ __forceinline__ v8f wmma_bf16(v16bf a, v16bf b, v8f c) {
  return __builtin_amdgcn_wmma_f32_16x16x32_bf16(false, a, false, b, (short)0, c,
                                                 false, false);
}

// --------------------------- workspace layout (bf16 elements) --------------
#define OFF_WCT   0u          // [32][192]
#define OFF_WI1T  6144u       // [400][1024]
#define OFF_WH1T  415744u     // [400][128]
#define OFF_WI2T  466944u     // [400][128]
#define OFF_WH2T  518144u     // [400][128]
#define OFF_WDT   569344u     // [32][96][128]
#define OFF_XPT   962560u     // [32][256][1024]
#define OFF_Z1_BYTES 18702336u // f32 [8192][400]

// ---------------------------------------------------------------------------
// Kernel 0: convert weights fp32 -> bf16, transposed [n][k], K zero-padded.
// ---------------------------------------------------------------------------
__global__ void __launch_bounds__(256)
prep_kernel(const float* __restrict__ conv_w, const float* __restrict__ Wi1,
            const float* __restrict__ Wh1, const float* __restrict__ Wi2,
            const float* __restrict__ Wh2, const float* __restrict__ dense_w,
            unsigned short* __restrict__ ws16)
{
  __bf16* Wct  = (__bf16*)(ws16 + OFF_WCT);
  __bf16* Wi1t = (__bf16*)(ws16 + OFF_WI1T);
  __bf16* Wh1t = (__bf16*)(ws16 + OFF_WH1T);
  __bf16* Wi2t = (__bf16*)(ws16 + OFF_WI2T);
  __bf16* Wh2t = (__bf16*)(ws16 + OFF_WH2T);
  __bf16* Wdt  = (__bf16*)(ws16 + OFF_WDT);

  int i = blockIdx.x * 256 + threadIdx.x;
  if (i < 32 * 192) {                       // conv: k = dl*64+ci, row = co
    int co = i / 192, k = i - co * 192;
    Wct[i] = f2bf(conv_w[k * 32 + co]);
  }
  if (i < 400 * 1024) {                     // Wi1t[n][k]
    int n = i >> 10, k = i & 1023;
    Wi1t[i] = f2bf(Wi1[k * 400 + n]);
  }
  if (i < 400 * 128) {                      // recurrent weights, K padded 100->128
    int n = i >> 7, k = i & 127;
    float a  = (k < HID) ? Wh1[k * 400 + n] : 0.f;
    float bb = (k < HID) ? Wi2[k * 400 + n] : 0.f;
    float cc = (k < HID) ? Wh2[k * 400 + n] : 0.f;
    Wh1t[i] = f2bf(a); Wi2t[i] = f2bf(bb); Wh2t[i] = f2bf(cc);
  }
  if (i < 32 * 96 * 128) {                  // dense slice per timestep: Wdt[t][n][u]
    int t = i / (96 * 128); int rem = i - t * 96 * 128;
    int n = rem >> 7, u = rem & 127;
    float v = (n < NCLS && u < HID) ? dense_w[(u * 32 + t) * NCLS + n] : 0.f;
    Wdt[i] = f2bf(v);
  }
}

// ---------------------------------------------------------------------------
// Kernel 1: conv1d(SAME,k=3) + bias + relu + avgpool2, implicit GEMM (K=192).
// Output transposed bf16: xp[c][b][lp].  Grid (8 l-blocks, 256 batch), 256 thr.
// ---------------------------------------------------------------------------
__global__ void __launch_bounds__(256)
conv_pool_kernel(const float* __restrict__ x, const float* __restrict__ conv_b,
                 const unsigned short* __restrict__ ws16,
                 unsigned short* __restrict__ xp16)
{
  const __bf16* Wct = (const __bf16*)(ws16 + OFF_WCT);
  __bf16* xp = (__bf16*)xp16;

  __shared__ __bf16 x_lds[258 * 72];    // rows l0-1 .. l0+256, stride 72 (conflict-free)
  __shared__ __bf16 pool_lds[32 * 130]; // [co][pooled pos], stride 130 (odd dwords)

  const int tid = threadIdx.x;
  const int lane = tid & 31, wave = tid >> 5;
  const int l15 = lane & 15, kg = lane >> 4;
  const int l0 = blockIdx.x * 256;
  const int b  = blockIdx.y;

  for (int i = tid; i < 258 * 64; i += 256) {
    int r = i >> 6, ci = i & 63;
    int l = l0 - 1 + r;
    float v = (l >= 0 && l < Lseq) ? x[((size_t)b * Lseq + l) * CIN + ci] : 0.f;
    x_lds[r * 72 + ci] = f2bf(v);
  }
  __syncthreads();

  float bias0 = conv_b[l15], bias1 = conv_b[16 + l15];
  for (int mt = 0; mt < 2; ++mt) {
    int pbase = wave * 32 + mt * 16;           // output position tile base
    v8f c0 = {}, c1 = {};
    for (int kc = 0; kc < 6; ++kc) {           // K = 192 = 6 * 32 (chunk = one tap half)
      int row = pbase + l15 + (kc >> 1);       // kc>>1 == tap dl
      int ci0 = (kc & 1) * 32 + kg * 16;
      v16bf a  = load_tileA(&x_lds[row * 72 + ci0]);
      v16bf b0 = load_tileB(&Wct[l15 * 192 + kc * 32 + kg * 16]);
      v16bf b1 = load_tileB(&Wct[(16 + l15) * 192 + kc * 32 + kg * 16]);
      c0 = wmma_bf16(a, b0, c0);
      c1 = wmma_bf16(a, b1, c1);
    }
    // bias + relu + pool: C rows M=r+8*kg; adjacent M pairs live in vgpr pairs.
    for (int q = 0; q < 4; ++q) {
      int prow = wave * 16 + mt * 8 + q + 4 * kg;  // pooled index within 128-block
      float a0 = fmaxf(c0[2 * q] + bias0, 0.f), a1 = fmaxf(c0[2 * q + 1] + bias0, 0.f);
      pool_lds[l15 * 130 + prow] = f2bf(0.5f * (a0 + a1));
      float d0 = fmaxf(c1[2 * q] + bias1, 0.f), d1 = fmaxf(c1[2 * q + 1] + bias1, 0.f);
      pool_lds[(16 + l15) * 130 + prow] = f2bf(0.5f * (d0 + d1));
    }
  }
  __syncthreads();
  int lp0 = blockIdx.x * 128;
  for (int i = tid; i < 32 * 128; i += 256) {   // coalesced transposed store
    int c = i >> 7, j = i & 127;
    xp[((size_t)c * Bsz + b) * LP + lp0 + j] = pool_lds[c * 130 + j];
  }
}

// ---------------------------------------------------------------------------
// Kernel 2: Z1[8192][400] = xp_t(8192x1024) @ Wi1(1024x400) + b1  (fp32 out).
// Grid (64 Mblocks, 5 Nchunks), 256 thr; WG tile 128x80, K streamed via LDS.
// ---------------------------------------------------------------------------
__global__ void __launch_bounds__(256)
gemm_zin_kernel(const unsigned short* __restrict__ xp16,
                const unsigned short* __restrict__ ws16,
                const float* __restrict__ b1, float* __restrict__ Z1)
{
  const __bf16* xp   = (const __bf16*)xp16;
  const __bf16* Wi1t = (const __bf16*)(ws16 + OFF_WI1T);

  __shared__ __bf16 As[128 * 40];  // stride 40 el = 20 dwords -> conflict-free b128
  __shared__ __bf16 Bs[80 * 40];

  const int tid = threadIdx.x;
  const int lane = tid & 31, wave = tid >> 5;
  const int l15 = lane & 15, kg = lane >> 4;
  const int Mbase = blockIdx.x * 128;
  const int Nbase = blockIdx.y * 80;
  const int arow = tid >> 1, ahalf = tid & 1;

  v8f C[5] = {};

  for (int kc = 0; kc < 32; ++kc) {
    __syncthreads();
    {  // stage A block 128x32 (each thread 32B)
      const __bf16* src = &xp[(size_t)(Mbase + arow) * LP + kc * 32 + ahalf * 16];
      __bf16* dst = &As[arow * 40 + ahalf * 16];
      ((u32x4*)dst)[0] = ((const u32x4*)src)[0];
      ((u32x4*)dst)[1] = ((const u32x4*)src)[1];
    }
    for (int i = tid; i < 320; i += 256) {  // stage B block 80x32
      int n = i >> 2, part = i & 3;
      *(u32x4*)&Bs[n * 40 + part * 8] =
          *(const u32x4*)&Wi1t[(size_t)(Nbase + n) * 1024 + kc * 32 + part * 8];
    }
    __syncthreads();
    v16bf a = load_tileA(&As[(wave * 16 + l15) * 40 + kg * 16]);
    for (int nt = 0; nt < 5; ++nt) {
      v16bf bm = load_tileB(&Bs[(nt * 16 + l15) * 40 + kg * 16]);
      C[nt] = wmma_bf16(a, bm, C[nt]);
    }
  }
  for (int nt = 0; nt < 5; ++nt) {
    int n = Nbase + nt * 16 + l15;
    float bias = b1[n];
    for (int r = 0; r < 8; ++r) {
      int row = Mbase + wave * 16 + r + 8 * kg;
      Z1[(size_t)row * NG + n] = C[nt][r] + bias;
    }
  }
}

// ---------------------------------------------------------------------------
// Kernel 3: fused LSTM1 + LSTM2 + dense.  16 WGs x 256 thr; WG g owns batch
// rows [16g,16g+16).  Recurrence is batch-independent -> no cross-WG sync.
// h kept as bf16 in LDS (K padded to 128); weights read from global (L2).
// ---------------------------------------------------------------------------
__global__ void __launch_bounds__(256)
lstm_fused_kernel(const float* __restrict__ Z1,
                  const unsigned short* __restrict__ ws16,
                  const float* __restrict__ b2,
                  const float* __restrict__ dense_b,
                  float* __restrict__ out)
{
  const __bf16* Wh1t = (const __bf16*)(ws16 + OFF_WH1T);
  const __bf16* Wi2t = (const __bf16*)(ws16 + OFF_WI2T);
  const __bf16* Wh2t = (const __bf16*)(ws16 + OFF_WH2T);
  const __bf16* Wdt  = (const __bf16*)(ws16 + OFF_WDT);

  __shared__ __bf16 h1bf[16 * 136];  // stride 136 el = 17x16B, odd-ish banks
  __shared__ __bf16 h2bf[16 * 136];
  __shared__ float  c1s[1600];
  __shared__ float  c2s[1600];
  __shared__ float  zs[16 * 400];    // gate pre-activations (z1 then z2)

  const int tid = threadIdx.x;
  const int lane = tid & 31, wave = tid >> 5;
  const int l15 = lane & 15, kg = lane >> 4;
  const int bg = blockIdx.x * 16;

  for (int i = tid; i < 16 * 136; i += 256) { h1bf[i] = f2bf(0.f); h2bf[i] = f2bf(0.f); }
  for (int i = tid; i < 1600; i += 256) { c1s[i] = 0.f; c2s[i] = 0.f; }
  __syncthreads();

  v8f accd = {};  // persistent dense accumulator (waves 0..5)

  for (int t = 0; t < 32; ++t) {
    // ---- z1 = Z1[t] + h1 @ Wh1 --------------------------------------------
    for (int nt = wave; nt < 25; nt += 8) {
      int n = nt * 16 + l15;
      v8f C;
      for (int r = 0; r < 8; ++r)
        C[r] = Z1[(size_t)(t * Bsz + bg + r + 8 * kg) * NG + n];
      for (int kc = 0; kc < 4; ++kc) {
        v16bf a  = load_tileA(&h1bf[l15 * 136 + kc * 32 + kg * 16]);
        v16bf bm = load_tileB(&Wh1t[n * 128 + kc * 32 + kg * 16]);
        C = wmma_bf16(a, bm, C);
      }
      for (int r = 0; r < 8; ++r)
        zs[(r + 8 * kg) * NG + n] = C[r];
    }
    __syncthreads();
    // ---- gates LSTM1 ------------------------------------------------------
    for (int e = tid; e < 1600; e += 256) {
      int bb = e / HID, u = e - bb * HID;
      const float* zr = &zs[bb * NG];
      float gi = zr[u], gf = zr[HID + u], gg = zr[2 * HID + u], go = zr[3 * HID + u];
      float c = sigm(gf) * c1s[e] + sigm(gi) * tanh_(gg);
      c1s[e] = c;
      h1bf[bb * 136 + u] = f2bf(sigm(go) * tanh_(c));
    }
    __syncthreads();
    // ---- z2 = b2 + h1 @ Wi2 + h2 @ Wh2 ------------------------------------
    for (int nt = wave; nt < 25; nt += 8) {
      int n = nt * 16 + l15;
      float bias = b2[n];
      v8f C;
      for (int r = 0; r < 8; ++r) C[r] = bias;
      for (int kc = 0; kc < 4; ++kc) {
        v16bf a  = load_tileA(&h1bf[l15 * 136 + kc * 32 + kg * 16]);
        v16bf bm = load_tileB(&Wi2t[n * 128 + kc * 32 + kg * 16]);
        C = wmma_bf16(a, bm, C);
      }
      for (int kc = 0; kc < 4; ++kc) {
        v16bf a  = load_tileA(&h2bf[l15 * 136 + kc * 32 + kg * 16]);
        v16bf bm = load_tileB(&Wh2t[n * 128 + kc * 32 + kg * 16]);
        C = wmma_bf16(a, bm, C);
      }
      for (int r = 0; r < 8; ++r)
        zs[(r + 8 * kg) * NG + n] = C[r];
    }
    __syncthreads();
    // ---- gates LSTM2 ------------------------------------------------------
    for (int e = tid; e < 1600; e += 256) {
      int bb = e / HID, u = e - bb * HID;
      const float* zr = &zs[bb * NG];
      float gi = zr[u], gf = zr[HID + u], gg = zr[2 * HID + u], go = zr[3 * HID + u];
      float c = sigm(gf) * c2s[e] + sigm(gi) * tanh_(gg);
      c2s[e] = c;
      h2bf[bb * 136 + u] = f2bf(sigm(go) * tanh_(c));
    }
    __syncthreads();
    // ---- dense partial: out += h2[t] @ dense_w[u*32+t, :] ------------------
    if (wave < 6) {
      for (int kc = 0; kc < 4; ++kc) {
        v16bf a  = load_tileA(&h2bf[l15 * 136 + kc * 32 + kg * 16]);
        v16bf bm = load_tileB(&Wdt[(t * 96 + wave * 16 + l15) * 128 + kc * 32 + kg * 16]);
        accd = wmma_bf16(a, bm, accd);
      }
    }
  }

  if (wave < 6) {
    int n = wave * 16 + l15;
    if (n < NCLS) {
      float db = dense_b[n];
      for (int r = 0; r < 8; ++r)
        out[(size_t)(bg + r + 8 * kg) * NCLS + n] = accd[r] + db;
    }
  }
}

// ---------------------------------------------------------------------------
extern "C" void kernel_launch(void* const* d_in, const int* in_sizes, int n_in,
                              void* d_out, int out_size, void* d_ws, size_t ws_size,
                              hipStream_t stream) {
  const float* x       = (const float*)d_in[0];
  const float* conv_w  = (const float*)d_in[1];
  const float* conv_b  = (const float*)d_in[2];
  const float* Wi1     = (const float*)d_in[3];
  const float* Wh1     = (const float*)d_in[4];
  const float* b1      = (const float*)d_in[5];
  const float* Wi2     = (const float*)d_in[6];
  const float* Wh2     = (const float*)d_in[7];
  const float* b2      = (const float*)d_in[8];
  const float* dense_w = (const float*)d_in[9];
  const float* dense_b = (const float*)d_in[10];
  float* out = (float*)d_out;

  unsigned short* ws16 = (unsigned short*)d_ws;
  unsigned short* xp16 = ws16 + OFF_XPT;
  float* Z1 = (float*)((char*)d_ws + OFF_Z1_BYTES);

  // 0) weight precompute (bf16, transposed, K-padded)
  prep_kernel<<<1600, 256, 0, stream>>>(conv_w, Wi1, Wh1, Wi2, Wh2, dense_w, ws16);
  // 1) conv + relu + pool -> xp_t[c][b][lp] bf16
  conv_pool_kernel<<<dim3(8, 256), 256, 0, stream>>>(x, conv_b, ws16, xp16);
  // 2) Z1 = xs @ Wi1 + b1 for all 32 timesteps at once
  gemm_zin_kernel<<<dim3(64, 5), 256, 0, stream>>>(xp16, ws16, b1, Z1);
  // 3) fused LSTM1 + LSTM2 + dense (batch-partitioned recurrence)
  lstm_fused_kernel<<<16, 256, 0, stream>>>(Z1, ws16, b2, dense_b, out);
}
